// InternalInteraction_65609920413983
// MI455X (gfx1250) — matrix operand order repaired and checked by reading, not compiled, not run on hardware
//
#include <hip/hip_runtime.h>

typedef __attribute__((ext_vector_type(16))) _Float16 v16h;
typedef __attribute__((ext_vector_type(8)))  _Float16 v8h;
typedef __attribute__((ext_vector_type(8)))  float    v8f;
typedef __attribute__((ext_vector_type(2)))  float    v2f;

#define BB   128
#define NN   32
#define DD   256
#define HH   1024
#define HT   64      // H-tile width per workgroup
#define XS   264     // xs row stride in halves (256 + pad, keeps 16B alignment)
#define WS   264     // wsT row stride in halves
#define HS   68      // hsum row stride in floats

// ---------------------------------------------------------------------------
// Kernel 1: hsum[b,j,h] = sum_i relu( (x[b,i,:]*x[b,j,:]) . W1[:,h] + b1[h] )
// One workgroup per (batch, 64-wide h tile). 256 threads = 8 waves.
// All m-tiles a wave touches share the same jbase -> accumulate post-ReLU
// partial sums in registers and flush with one LDS-atomic pass at the end.
// ---------------------------------------------------------------------------
__global__ __launch_bounds__(256)
void interaction_mlp1(const float* __restrict__ x,
                      const float* __restrict__ W1,
                      const float* __restrict__ b1,
                      float* __restrict__ hsum_g) {
  __shared__ __align__(16) _Float16 xs [NN * XS];   // x[b] as f16, 32 x 256
  __shared__ __align__(16) _Float16 wsT[HT * WS];   // wsT[n][k] = W1[k][h0+n], f16
  __shared__ float hsum[NN * HS];                   // 32 x 64 f32 reduction tile

  const int tid  = threadIdx.x;
  const int wave = tid >> 5;
  const int lane = tid & 31;
  const int l16  = lane & 15;
  const int hi   = lane >> 4;

  const int b  = blockIdx.x >> 4;          // 16 h-tiles per batch
  const int h0 = (blockIdx.x & 15) * HT;

  // stage x[b] -> f16 LDS
  for (int e = tid; e < NN * DD; e += 256) {
    int r = e >> 8, k = e & 255;
    xs[r * XS + k] = (_Float16)x[b * (NN * DD) + e];
  }
  // stage W1[:, h0:h0+64] transposed -> f16 LDS (k contiguous per column)
  for (int e = tid; e < DD * HT; e += 256) {
    int k = e >> 6, n = e & 63;
    wsT[n * WS + k] = (_Float16)W1[k * HH + h0 + n];
  }
  for (int e = tid; e < NN * HS; e += 256) hsum[e] = 0.0f;
  __syncthreads();

  // bias for the 4 n-tiles handled by this lane's column
  float bias[4];
#pragma unroll
  for (int nt = 0; nt < 4; ++nt) bias[nt] = b1[h0 + nt * 16 + l16];

  // Register accumulator of post-ReLU values over all i handled by this wave.
  // Rows (j) covered are identical for every m-tile this wave processes.
  float facc[4][8];
#pragma unroll
  for (int nt = 0; nt < 4; ++nt)
#pragma unroll
    for (int v = 0; v < 8; ++v) facc[nt][v] = 0.0f;

  // 64 M-tiles of 16 rows ((i,j) pairs); wave-cyclic distribution
  for (int mt = wave; mt < 64; mt += 8) {
    const int m0 = mt * 16;
    const int r  = m0 + l16;        // this lane's A-matrix row (i*32 + j)
    const _Float16* xi = &xs[(r >> 5) * XS];
    const _Float16* xj = &xs[(r & 31) * XS];

    v8f acc[4];
#pragma unroll
    for (int nt = 0; nt < 4; ++nt) acc[nt] = (v8f){0.f,0.f,0.f,0.f,0.f,0.f,0.f,0.f};

#pragma unroll
    for (int c = 0; c < 8; ++c) {          // K = 256 in chunks of 32
      // A layout (16-bit 16x32): lane hi-half selects K groups 8..15 / 24..31
      const int ka = c * 32 + hi * 8;
      v8h alo = *(const v8h*)(xi + ka)      * *(const v8h*)(xj + ka);
      v8h ahi = *(const v8h*)(xi + ka + 16) * *(const v8h*)(xj + ka + 16);
      v16h A = __builtin_shufflevector(alo, ahi,
                 0,1,2,3,4,5,6,7,8,9,10,11,12,13,14,15);
      // B layout (16-bit 32x16): column = l16, lane hi-half selects K groups 8..15 / 24..31
      const int kb = c * 32 + hi * 8;
#pragma unroll
      for (int nt = 0; nt < 4; ++nt) {
        const _Float16* wrow = &wsT[(nt * 16 + l16) * WS];
        v8h blo = *(const v8h*)(wrow + kb);
        v8h bhi = *(const v8h*)(wrow + kb + 16);
        v16h Bf = __builtin_shufflevector(blo, bhi,
                    0,1,2,3,4,5,6,7,8,9,10,11,12,13,14,15);
        acc[nt] = __builtin_amdgcn_wmma_f32_16x16x32_f16(
            false, A, false, Bf, (short)0, acc[nt], false, false);
      }
    }

    // bias + relu, accumulate over this m-tile's i in registers
#pragma unroll
    for (int nt = 0; nt < 4; ++nt) {
#pragma unroll
      for (int v = 0; v < 8; ++v) {
        float hv = acc[nt][v] + bias[nt];
        facc[nt][v] += (hv > 0.0f ? hv : 0.0f);
      }
    }
  }

  // single LDS-atomic flush (4 waves collide per row set -> ds_add_f32)
  {
    const int jbase = (wave & 1) * 16 + 8 * hi;  // j of element v is jbase+v
#pragma unroll
    for (int nt = 0; nt < 4; ++nt) {
      const int col = nt * 16 + l16;
#pragma unroll
      for (int v = 0; v < 8; ++v) {
        atomicAdd(&hsum[(jbase + v) * HS + col], facc[nt][v]);
      }
    }
  }
  __syncthreads();

  // write hsum tile to global scratch
  for (int e = tid; e < NN * HT; e += 256) {
    int j = e >> 6, n = e & 63;
    hsum_g[(b * NN + j) * HH + h0 + n] = hsum[j * HS + n];
  }
}

// ---------------------------------------------------------------------------
// Kernel 2: out[row, d] = hsum[row, :] . W2[:, d] + 32*b2[d]   (rows = b*32+j)
// Full-precision f32 WMMA (16x16x4). One wave per 16x16 output tile.
// ---------------------------------------------------------------------------
__global__ __launch_bounds__(256)
void interaction_mlp2(const float* __restrict__ hsum_g,
                      const float* __restrict__ W2,
                      const float* __restrict__ b2,
                      float* __restrict__ out) {
  const int tid  = threadIdx.x;
  const int wave = tid >> 5;
  const int lane = tid & 31;
  const int l16  = lane & 15;
  const int hi   = lane >> 4;

  const int tile = blockIdx.x * 8 + wave;  // 4096 tiles = 256 mt x 16 nt
  const int m0 = (tile >> 4) * 16;
  const int n0 = (tile & 15) * 16;

  const float* arow = hsum_g + (size_t)(m0 + l16) * HH;
  v8f acc = (v8f){0.f,0.f,0.f,0.f,0.f,0.f,0.f,0.f};

  for (int k0 = 0; k0 < HH; k0 += 4) {
    const int ka = k0 + hi * 2;            // f32 A 16x4: lanes 16-31 hold K=2,3
    v2f a = *(const v2f*)(arow + ka);
    v2f bv;
    bv.x = W2[(size_t)ka * DD + n0 + l16];
    bv.y = W2[(size_t)(ka + 1) * DD + n0 + l16];
    acc = __builtin_amdgcn_wmma_f32_16x16x4_f32(
        false, a, false, bv, (short)0, acc, false, false);
  }

  const int col = n0 + l16;
  const float bb = 32.0f * b2[col];
#pragma unroll
  for (int v = 0; v < 8; ++v) {
    const int row = m0 + v + 8 * hi;
    out[(size_t)row * DD + col] = acc[v] + bb;
  }
}

// ---------------------------------------------------------------------------
extern "C" void kernel_launch(void* const* d_in, const int* in_sizes, int n_in,
                              void* d_out, int out_size, void* d_ws, size_t ws_size,
                              hipStream_t stream) {
  const float* x  = (const float*)d_in[0];   // [128,32,256]
  const float* W1 = (const float*)d_in[1];   // [256,1024]
  const float* b1 = (const float*)d_in[2];   // [1024]
  const float* W2 = (const float*)d_in[3];   // [1024,256]
  const float* b2 = (const float*)d_in[4];   // [256]
  float* out    = (float*)d_out;             // [128,32,256]
  float* hsum_g = (float*)d_ws;              // [128*32, 1024] f32 = 16 MB scratch

  // Kernel 1: 128 batches x 16 h-tiles
  interaction_mlp1<<<BB * 16, 256, 0, stream>>>(x, W1, b1, hsum_g);
  // Kernel 2: 4096 tiles / 8 waves per block
  interaction_mlp2<<<512, 256, 0, stream>>>(hsum_g, W2, b2, out);
}
